// Scaled_Dot_Product_Attention_18391049961538
// MI455X (gfx1250) — matrix operand order repaired and checked by reading, not compiled, not run on hardware
//
#include <hip/hip_runtime.h>
#include <math.h>

// Problem constants from the reference: B=4, H=16, S=2048, D=64
#define B_ 4
#define H_ 16
#define S_ 2048
#define D_ 64

typedef __attribute__((ext_vector_type(16))) __bf16 v16bf;
typedef __attribute__((ext_vector_type(8)))  float  v8f;

union Frag { unsigned int u[8]; uint4 q[2]; v16bf v; };

__device__ __forceinline__ unsigned short f2bf(float f) {
    union { float f; unsigned int u; } c; c.f = f;
    unsigned int u = c.u + 0x7FFFu + ((c.u >> 16) & 1u);  // round-to-nearest-even
    return (unsigned short)(u >> 16);
}
__device__ __forceinline__ unsigned int pack2(float lo, float hi) {
    return (unsigned int)f2bf(lo) | ((unsigned int)f2bf(hi) << 16);
}

__global__ __launch_bounds__(256)
void Scaled_Dot_Product_Attention_18391049961538_kernel(
    const float* __restrict__ Q, const float* __restrict__ K,
    const float* __restrict__ V, const unsigned char* __restrict__ M,
    float* __restrict__ O)
{
    // LDS staging: K^T[d][k] bf16, V[k][d] bf16, mask tile, per-wave S tiles
    __shared__ __align__(16) unsigned short sKt[D_ * 32];      // 4 KB
    __shared__ __align__(16) unsigned short sV [32 * D_];      // 4 KB
    __shared__ __align__(16) unsigned char  sM [128 * 32];     // 4 KB
    __shared__ __align__(16) unsigned short sS [8][16 * 32];   // 8 KB

    const int tid  = threadIdx.x;
    const int wave = tid >> 5;
    const int lane = tid & 31;
    const int qblk = blockIdx.x, h = blockIdx.y, b = blockIdx.z;
    const int q0   = qblk * 128;

    const size_t headoff = ((size_t)b * H_ + h) * (size_t)S_ * D_;
    const float* Qh = Q + headoff;
    const float* Kh = K + headoff;
    const float* Vh = V + headoff;
    const unsigned char* Mh = M + (size_t)b * S_ * S_;   // mask broadcast over heads

    // ---- Q A-fragments (16x32 bf16 x2 covering D=64), scaled by 1/sqrt(64) ----
    Frag aq[2];
    {
        const int   qrow = q0 + wave * 16 + (lane & 15);
        const float* qp  = Qh + (size_t)qrow * D_;
        const int   lg   = (lane & 16) ? 8 : 0;   // lanes>=16 hold K=8..15 / 24..31
        const float s    = 0.125f;
#pragma unroll
        for (int c = 0; c < 2; ++c) {
            const int ds0 = 32 * c + lg;
            float4 f0 = *(const float4*)(qp + ds0);
            float4 f1 = *(const float4*)(qp + ds0 + 4);
            float4 f2 = *(const float4*)(qp + ds0 + 16);
            float4 f3 = *(const float4*)(qp + ds0 + 20);
            aq[c].u[0] = pack2(f0.x * s, f0.y * s);
            aq[c].u[1] = pack2(f0.z * s, f0.w * s);
            aq[c].u[2] = pack2(f1.x * s, f1.y * s);
            aq[c].u[3] = pack2(f1.z * s, f1.w * s);
            aq[c].u[4] = pack2(f2.x * s, f2.y * s);
            aq[c].u[5] = pack2(f2.z * s, f2.w * s);
            aq[c].u[6] = pack2(f3.x * s, f3.y * s);
            aq[c].u[7] = pack2(f3.z * s, f3.w * s);
        }
    }

    v8f accO[4];
#pragma unroll
    for (int d = 0; d < 4; ++d) accO[d] = {};

    const int krow  = tid >> 3;        // 0..31 : k row staged by this thread
    const int d0    = (tid & 7) * 8;   // 8 d-elements per thread
    const int mq    = tid >> 1;        // 0..127: mask row staged
    const int mhalf = tid & 1;

    // Per-thread operands for the async mask DMA (GVS addressing:
    //   global = SADDR(64) + VADDR(i32); LDS dest from VDST vgpr).
    // Low 32 bits of a generic pointer to __shared__ == LDS byte offset.
    const unsigned mask_lds  = (unsigned)(uintptr_t)&sM[mq * 32 + mhalf * 16];
    const unsigned mask_voff = (unsigned)((q0 + mq) * S_ + mhalf * 16);

    for (int k0 = 0; k0 < S_; k0 += 32) {
        __syncthreads();
        // ---- async DMA: mask tile bytes straight into LDS (no conversion) ----
        {
            const unsigned long long mbase = (unsigned long long)(uintptr_t)(Mh + k0);
            asm volatile("global_load_async_to_lds_b128 %0, %1, %2"
                         :: "v"(mask_lds), "v"(mask_voff), "s"(mbase)
                         : "memory");
        }
        // ---- cooperative staging: K^T, V (fp32 -> bf16) ----
        {
            const float* kp = Kh + (size_t)(k0 + krow) * D_ + d0;
            float4 f0 = *(const float4*)(kp);
            float4 f1 = *(const float4*)(kp + 4);
            unsigned short hv[8] = { f2bf(f0.x), f2bf(f0.y), f2bf(f0.z), f2bf(f0.w),
                                     f2bf(f1.x), f2bf(f1.y), f2bf(f1.z), f2bf(f1.w) };
#pragma unroll
            for (int j = 0; j < 8; ++j) sKt[(d0 + j) * 32 + krow] = hv[j];

            const float* vp = Vh + (size_t)(k0 + krow) * D_ + d0;
            float4 g0 = *(const float4*)(vp);
            float4 g1 = *(const float4*)(vp + 4);
            uint4 pv;
            pv.x = pack2(g0.x, g0.y); pv.y = pack2(g0.z, g0.w);
            pv.z = pack2(g1.x, g1.y); pv.w = pack2(g1.z, g1.w);
            *(uint4*)&sV[krow * D_ + d0] = pv;

            // prefetch next tile's K/V rows into cache while we compute
            if (k0 + 32 < S_) {
                __builtin_prefetch(Kh + (size_t)(k0 + 32 + krow) * D_ + d0, 0, 0);
                __builtin_prefetch(Vh + (size_t)(k0 + 32 + krow) * D_ + d0, 0, 0);
            }
        }
        // drain this wave's async copies so LDS is valid past the barrier
        asm volatile("s_wait_asynccnt 0" ::: "memory");
        __syncthreads();

        // ---- S = (Q/8) @ K^T : two 16x16 f32 tiles, mask, spill as bf16 ----
        const int hh   = lane >> 4;
        const int ncol = lane & 15;
#pragma unroll
        for (int ksub = 0; ksub < 2; ++ksub) {
            v8f accS = {};
#pragma unroll
            for (int c = 0; c < 2; ++c) {
                Frag bk;
                const int kc = ksub * 16 + ncol;
#pragma unroll
                for (int j = 0; j < 8; ++j) {
                    const int dd = 32 * c + 8 * hh + 2 * j + (j >= 4 ? 8 : 0);
                    bk.u[j] = (unsigned int)sKt[dd * 32 + kc] | ((unsigned int)sKt[(dd + 1) * 32 + kc] << 16);
                }
                accS = __builtin_amdgcn_wmma_f32_16x16x32_bf16(
                    false, aq[c].v, false, bk.v, (short)0, accS, false, false);
            }
            const int nloc = ksub * 16 + ncol;
#pragma unroll
            for (int v = 0; v < 8; ++v) {
                const int mrow = v + 8 * hh;                 // C-layout row
                const int ql   = wave * 16 + mrow;           // local q (0..127)
                float val = sM[ql * 32 + nloc] ? -1e9f : accS[v];
                sS[wave][mrow * 32 + nloc] = f2bf(val);
            }
        }

        // ---- O += S @ V over all four 16-wide d sub-tiles ----
        {
            const int m   = lane & 15;
            const int g0i = (lane & 16) ? 8 : 0;
            Frag as;
            as.q[0] = *(const uint4*)&sS[wave][m * 32 + g0i];
            as.q[1] = *(const uint4*)&sS[wave][m * 32 + 16 + g0i];
#pragma unroll
            for (int dsub = 0; dsub < 4; ++dsub) {
                Frag bv;
#pragma unroll
                for (int j = 0; j < 8; ++j) {
                    const int kk = g0i + 2 * j + (j >= 4 ? 8 : 0);
                    bv.u[j] = (unsigned int)sV[kk * D_ + dsub * 16 + m] | ((unsigned int)sV[(kk + 1) * D_ + dsub * 16 + m] << 16);
                }
                accO[dsub] = __builtin_amdgcn_wmma_f32_16x16x32_bf16(
                    false, as.v, false, bv.v, (short)0, accO[dsub], false, false);
            }
        }
    }

    // ---- log_softmax over D=64 per query row, then store f32 ----
    float* Oh = O + headoff + (size_t)(q0 + wave * 16) * D_;
    const int ncol = lane & 15;
    const int hh   = lane >> 4;
#pragma unroll
    for (int v = 0; v < 8; ++v) {
        const int row = v + 8 * hh;
        float x0 = accO[0][v], x1 = accO[1][v], x2 = accO[2][v], x3 = accO[3][v];
        float rmax = fmaxf(fmaxf(x0, x1), fmaxf(x2, x3));
#pragma unroll
        for (int mxy = 1; mxy <= 8; mxy <<= 1)
            rmax = fmaxf(rmax, __shfl_xor(rmax, mxy, 32));   // 16-lane row group
        float ssum = expf(x0 - rmax) + expf(x1 - rmax) + expf(x2 - rmax) + expf(x3 - rmax);
#pragma unroll
        for (int mxy = 1; mxy <= 8; mxy <<= 1)
            ssum += __shfl_xor(ssum, mxy, 32);
        float lse = rmax + logf(ssum);
        float* orow = Oh + (size_t)row * D_ + ncol;
        orow[0]  = x0 - lse;
        orow[16] = x1 - lse;
        orow[32] = x2 - lse;
        orow[48] = x3 - lse;
    }
}

extern "C" void kernel_launch(void* const* d_in, const int* in_sizes, int n_in,
                              void* d_out, int out_size, void* d_ws, size_t ws_size,
                              hipStream_t stream) {
    const float* Q = (const float*)d_in[0];
    const float* K = (const float*)d_in[1];
    const float* V = (const float*)d_in[2];
    const unsigned char* M = (const unsigned char*)d_in[3];  // jax bool mask, 1 byte/elem
    float* O = (float*)d_out;

    dim3 grid(S_ / 128, H_, B_);   // 16 q-blocks x 16 heads x 4 batches
    Scaled_Dot_Product_Attention_18391049961538_kernel<<<grid, 256, 0, stream>>>(Q, K, V, M, O);
}
